// SimplicialConv_5342939316461
// MI455X (gfx1250) — matrix operand order repaired and checked by reading, not compiled, not run on hardware
//
#include <hip/hip_runtime.h>
#include <hip/hip_bf16.h>

// ---------------------------------------------------------------------------
// SimplicialConv (two 3-term Chebyshev stacks + dense contraction) for gfx1250
//
// Pipeline:
//   1) transpose x (B,C,M) -> T0 node-major [m][b*32+c]   (coalesced SpMM I/O)
//   2) init: Tl1=Tu1=0 ; Tl2=Tu2=-T0                      (folds "-T_{k-2}")
//   3) 4x SpMM scatter-add: dst[row][:] += scale*val*src[col][:]
//      one wave32 per edge, 4 channels/lane, f32 atomics resolving in L2
//   4) GEMM (B*M)x192 @ 192x32 via v_wmma_f32_16x16x32_bf16, bf16x3 split
//      for ~fp32 accuracy; K=192 == 6 terms x 32 channels (one WMMA K-step
//      per Chebyshev term). Bias added in epilogue.
// ---------------------------------------------------------------------------

typedef __attribute__((ext_vector_type(16))) __bf16 v16bf;
typedef __attribute__((ext_vector_type(8)))  float  v8f;

#define BDIM   4
#define CIN    32
#define COUT   32
#define KTERMS 6
#define NCH    128   // B*C? no: channels per node = BDIM*CIN? -> see note below

// NOTE: node-major feature vector per node m is 128 floats indexed b*32+c.

// --------------------------- 1) transpose ---------------------------------
__global__ __launch_bounds__(256) void transpose_kernel(
    const float* __restrict__ x, float* __restrict__ t0, int Mlen)
{
  __shared__ float tile[32][33];
  const int b  = blockIdx.y;
  const int m0 = blockIdx.x * 32;
  const int tx = threadIdx.x;       // 0..31
  const int ty = threadIdx.y;       // 0..7
#pragma unroll
  for (int j = 0; j < 4; ++j) {
    const int c = ty + j * 8;
    const int m = m0 + tx;
    tile[c][tx] = (m < Mlen) ? x[((size_t)(b * CIN + c)) * Mlen + m] : 0.f;
  }
  __syncthreads();
#pragma unroll
  for (int j = 0; j < 4; ++j) {
    const int mm = ty + j * 8;
    const int m  = m0 + mm;
    if (m < Mlen)
      t0[(size_t)m * 128 + b * CIN + tx] = tile[tx][mm];
  }
}

// --------------------------- 2) init terms --------------------------------
__global__ __launch_bounds__(256) void init_terms_kernel(
    const float* __restrict__ t0,
    float* __restrict__ tl1, float* __restrict__ tl2,
    float* __restrict__ tu1, float* __restrict__ tu2, size_t n)
{
  const size_t i = (size_t)blockIdx.x * blockDim.x + threadIdx.x;
  if (i < n) {
    const float neg = -t0[i];
    tl1[i] = 0.f;
    tu1[i] = 0.f;
    tl2[i] = neg;
    tu2[i] = neg;
  }
}

// --------------------------- 3) SpMM scatter ------------------------------
// One wave per edge: lane l handles channels 4l..4l+3 (512B coalesced).
__global__ __launch_bounds__(256) void spmm_kernel(
    const int* __restrict__ rows, const int* __restrict__ cols,
    const float* __restrict__ vals, const float* __restrict__ src,
    float* __restrict__ dst, float scale, int nnz)
{
  const int edge = (int)(blockIdx.x * (blockDim.x >> 5)) + (threadIdx.x >> 5);
  const int lane = threadIdx.x & 31;
  if (edge >= nnz) return;
  const int   r = rows[edge];         // wave-uniform -> scalar loads
  const int   c = cols[edge];
  const float v = vals[edge] * scale;
  const float4 s = *reinterpret_cast<const float4*>(src + (size_t)c * 128 + lane * 4);
  float* d = dst + (size_t)r * 128 + lane * 4;
  __hip_atomic_fetch_add(d + 0, v * s.x, __ATOMIC_RELAXED, __HIP_MEMORY_SCOPE_AGENT);
  __hip_atomic_fetch_add(d + 1, v * s.y, __ATOMIC_RELAXED, __HIP_MEMORY_SCOPE_AGENT);
  __hip_atomic_fetch_add(d + 2, v * s.z, __ATOMIC_RELAXED, __HIP_MEMORY_SCOPE_AGENT);
  __hip_atomic_fetch_add(d + 3, v * s.w, __ATOMIC_RELAXED, __HIP_MEMORY_SCOPE_AGENT);
}

// --------------------------- 4) WMMA GEMM ---------------------------------
// Per wave: one 16(m) x 32(o) output tile for a fixed batch b.
// K loop = 6 terms; per term one 16x32 A-fragment (bf16 hi/lo) and two
// 32x16 B-fragments (theta slice, bf16 hi/lo) -> 6 WMMAs (bf16x3 x 2 tiles).
__global__ __launch_bounds__(256) void cheb_gemm_kernel(
    const float* __restrict__ t0,  const float* __restrict__ tl1,
    const float* __restrict__ tl2, const float* __restrict__ tu1,
    const float* __restrict__ tu2, const float* __restrict__ theta,
    const float* __restrict__ bias, float* __restrict__ out, int Mlen)
{
  __shared__ __bf16 thetaHi[KTERMS * CIN * COUT];   // [k][i][o]
  __shared__ __bf16 thetaLo[KTERMS * CIN * COUT];

  const int tid = threadIdx.x;
  for (int idx = tid; idx < KTERMS * CIN * COUT; idx += 256) {
    const int o = idx & 31;
    const int i = (idx >> 5) & 31;
    const int k = idx >> 10;
    const float  v = theta[(size_t)o * (CIN * KTERMS) + i * KTERMS + k];
    const __bf16 h = (__bf16)v;
    thetaHi[idx] = h;
    thetaLo[idx] = (__bf16)(v - (float)h);
  }
  __syncthreads();

  const int lane = tid & 31;
  const int wave = tid >> 5;
  const int b    = blockIdx.y;
  const int m0   = (int)blockIdx.x * 128 + wave * 16;
  if (m0 >= Mlen) return;                 // wave-uniform; EXEC stays all-1s

  const float* terms[KTERMS] = { t0, tl1, tl2, t0, tu1, tu2 };

  v8f acc0 = {};                          // o = 0..15
  v8f acc1 = {};                          // o = 16..31
  const int row = lane & 15;              // A: lane -> M row (ISA 7.12.2)
  const int hi8 = (lane >> 4) * 8;        // lane-group K offset / D row offset

#pragma unroll
  for (int k = 0; k < KTERMS; ++k) {
    // ---- A fragment: rows m0+row, reduction index i = channel ----
    const float* rowp = terms[k] + (size_t)(m0 + row) * 128 + b * CIN;
    v16bf aHi, aLo;
#pragma unroll
    for (int p = 0; p < 8; ++p) {
      const float  v0 = rowp[hi8 + p];            // K = hi8+p
      const __bf16 h0 = (__bf16)v0;
      aHi[p]     = h0;
      aLo[p]     = (__bf16)(v0 - (float)h0);
      const float  v1 = rowp[16 + hi8 + p];       // K = 16+hi8+p
      const __bf16 h1 = (__bf16)v1;
      aHi[8 + p] = h1;
      aLo[8 + p] = (__bf16)(v1 - (float)h1);
    }
    // ---- B fragments: lane&15 -> N (=o), elements -> K (=i) in A's order ----
    const __bf16* bpHi = &thetaHi[k * 1024 + row];
    const __bf16* bpLo = &thetaLo[k * 1024 + row];
    v16bf b0Hi, b0Lo, b1Hi, b1Lo;
#pragma unroll
    for (int p = 0; p < 8; ++p) {
      const int i0 = (hi8 + p) * 32, i1 = (16 + hi8 + p) * 32;
      b0Hi[p] = bpHi[i0];          b0Lo[p] = bpLo[i0];           // o = 0..15
      b0Hi[8 + p] = bpHi[i1];      b0Lo[8 + p] = bpLo[i1];
      b1Hi[p] = bpHi[16 + i0];     b1Lo[p] = bpLo[16 + i0];      // o = 16..31
      b1Hi[8 + p] = bpHi[16 + i1]; b1Lo[8 + p] = bpLo[16 + i1];
    }
    // bf16x3 split-precision accumulate (hi*hi + hi*lo + lo*hi)
    acc0 = __builtin_amdgcn_wmma_f32_16x16x32_bf16(false, aHi, false, b0Hi, (short)0, acc0, false, false);
    acc0 = __builtin_amdgcn_wmma_f32_16x16x32_bf16(false, aHi, false, b0Lo, (short)0, acc0, false, false);
    acc0 = __builtin_amdgcn_wmma_f32_16x16x32_bf16(false, aLo, false, b0Hi, (short)0, acc0, false, false);
    acc1 = __builtin_amdgcn_wmma_f32_16x16x32_bf16(false, aHi, false, b1Hi, (short)0, acc1, false, false);
    acc1 = __builtin_amdgcn_wmma_f32_16x16x32_bf16(false, aHi, false, b1Lo, (short)0, acc1, false, false);
    acc1 = __builtin_amdgcn_wmma_f32_16x16x32_bf16(false, aLo, false, b1Hi, (short)0, acc1, false, false);
  }

  // ---- Epilogue: D VGPR r -> M = r (+8 for upper lane group), N = lane&15
  const int   ocol  = lane & 15;
  const float bias0 = bias[ocol];
  const float bias1 = bias[16 + ocol];
#pragma unroll
  for (int r = 0; r < 8; ++r) {
    const int m = m0 + r + hi8;
    out[(size_t)(b * COUT + ocol)      * Mlen + m] = acc0[r] + bias0;
    out[(size_t)(b * COUT + 16 + ocol) * Mlen + m] = acc1[r] + bias1;
  }
}

// --------------------------- launcher -------------------------------------
extern "C" void kernel_launch(void* const* d_in, const int* in_sizes, int n_in,
                              void* d_out, int out_size, void* d_ws, size_t ws_size,
                              hipStream_t stream) {
  const float* x       = (const float*)d_in[0];
  const int*   Ll_rows = (const int*)  d_in[1];
  const int*   Ll_cols = (const int*)  d_in[2];
  const float* Ll_vals = (const float*)d_in[3];
  const int*   Lu_rows = (const int*)  d_in[4];
  const int*   Lu_cols = (const int*)  d_in[5];
  const float* Lu_vals = (const float*)d_in[6];
  const float* theta   = (const float*)d_in[7];
  const float* bias    = (const float*)d_in[8];
  float*       out     = (float*)d_out;

  const int Mlen = in_sizes[0] / (BDIM * CIN);
  const int nnzL = in_sizes[1];
  const int nnzU = in_sizes[4];

  const size_t tsz = (size_t)Mlen * 128;       // floats per term buffer
  float* t0  = (float*)d_ws;
  float* tl1 = t0  + tsz;
  float* tl2 = tl1 + tsz;
  float* tu1 = tl2 + tsz;
  float* tu2 = tu1 + tsz;

  // 1) transpose x -> node-major T0
  {
    dim3 tb(32, 8);
    dim3 tg((Mlen + 31) / 32, BDIM);
    transpose_kernel<<<tg, tb, 0, stream>>>(x, t0, Mlen);
  }
  // 2) init T1 buffers to 0, T2 buffers to -T0 (every call: no stale state)
  {
    const size_t n = tsz;
    init_terms_kernel<<<(unsigned)((n + 255) / 256), 256, 0, stream>>>(
        t0, tl1, tl2, tu1, tu2, n);
  }
  // 3) four SpMMs: T1 = L*T0 ; T2 += 2*L*T1 (T2 pre-seeded with -T0)
  spmm_kernel<<<(nnzL + 7) / 8, 256, 0, stream>>>(Ll_rows, Ll_cols, Ll_vals, t0,  tl1, 1.0f, nnzL);
  spmm_kernel<<<(nnzU + 7) / 8, 256, 0, stream>>>(Lu_rows, Lu_cols, Lu_vals, t0,  tu1, 1.0f, nnzU);
  spmm_kernel<<<(nnzL + 7) / 8, 256, 0, stream>>>(Ll_rows, Ll_cols, Ll_vals, tl1, tl2, 2.0f, nnzL);
  spmm_kernel<<<(nnzU + 7) / 8, 256, 0, stream>>>(Lu_rows, Lu_cols, Lu_vals, tu1, tu2, 2.0f, nnzU);
  // 4) dense contraction with WMMA + bias
  {
    dim3 gg((Mlen + 127) / 128, BDIM);
    cheb_gemm_kernel<<<gg, 256, 0, stream>>>(t0, tl1, tl2, tu1, tu2,
                                             theta, bias, out, Mlen);
  }
}
